// DPG_Predictor_31748398252687
// MI455X (gfx1250) — matrix operand order repaired and checked by reading, not compiled, not run on hardware
//
#include <hip/hip_runtime.h>

// Problem constants (from reference)
#define NN   2048   // graph nodes
#define BQ   8      // batch
#define CM   32     // d_model
#define DFF  256    // d_ff
#define T0   13     // input time length
#define NCOL 16     // columns per block in final fused kernel

typedef __attribute__((ext_vector_type(16))) __bf16 v16bf;
typedef __attribute__((ext_vector_type(8)))  __bf16 v8bf;
typedef __attribute__((ext_vector_type(8)))  float  v8f;

__device__ __forceinline__ __bf16 f2bf(float f) {
  // round-to-nearest-even f32 -> bf16
  union { unsigned short s; __bf16 b; } u;
  unsigned int x = __float_as_uint(f);
  u.s = (unsigned short)((x + 0x7FFFu + ((x >> 16) & 1u)) >> 16);
  return u.b;
}

// ---------------------------------------------------------------------------
// Adjacency: f32 [b][v][w] -> bf16 transposed [b][w][v]  (LDS-tiled)
// ---------------------------------------------------------------------------
__global__ __launch_bounds__(256) void k_adj_t_bf16(const float* __restrict__ A,
                                                    __bf16* __restrict__ AT) {
  __shared__ float tile[32][33];
  const int b  = blockIdx.z;
  const int v0 = blockIdx.y * 32, w0 = blockIdx.x * 32;
  const int tx = threadIdx.x & 31, ty = threadIdx.x >> 5;  // 32 x 8
  const float*  Ab  = A  + (size_t)b * NN * NN;
  __bf16*       ATb = AT + (size_t)b * NN * NN;
#pragma unroll
  for (int i = 0; i < 32; i += 8)
    tile[ty + i][tx] = Ab[(size_t)(v0 + ty + i) * NN + (w0 + tx)];
  __syncthreads();
#pragma unroll
  for (int i = 0; i < 32; i += 8)
    ATb[(size_t)(w0 + ty + i) * NN + (v0 + tx)] = f2bf(tile[tx][ty + i]);
}

// ---------------------------------------------------------------------------
// start conv1x1: x [B][2][N][13] -> y [B][32][N][13]
// ---------------------------------------------------------------------------
__global__ void k_start(const float* __restrict__ x, const float* __restrict__ W,
                        const float* __restrict__ bias, float* __restrict__ y) {
  size_t idx = (size_t)blockIdx.x * blockDim.x + threadIdx.x;
  const size_t total = (size_t)BQ * CM * NN * T0;
  if (idx >= total) return;
  size_t nt = idx % ((size_t)NN * T0);
  size_t bo = idx / ((size_t)NN * T0);
  int o = (int)(bo % CM);
  int b = (int)(bo / CM);
  const float* xb = x + (size_t)b * 2 * NN * T0 + nt;
  y[idx] = W[o * 2] * xb[0] + W[o * 2 + 1] * xb[(size_t)NN * T0] + bias[o];
}

// ---------------------------------------------------------------------------
// filter/gate conv1x2 + tanh*sigmoid:  x [B][32][N][Tin] -> xg [B][32][N][Tin-1]
// ---------------------------------------------------------------------------
__global__ void k_filt_gate(const float* __restrict__ x,
                            const float* __restrict__ fw, const float* __restrict__ fb,
                            const float* __restrict__ gw, const float* __restrict__ gb,
                            float* __restrict__ xg, int Tin) {
  const int Tout = Tin - 1;
  size_t idx = (size_t)blockIdx.x * blockDim.x + threadIdx.x;
  const size_t total = (size_t)BQ * CM * NN * Tout;
  if (idx >= total) return;
  int t = (int)(idx % Tout);
  size_t r = idx / Tout;
  int n = (int)(r % NN);
  size_t bo = r / NN;
  int o = (int)(bo % CM);
  int b = (int)(bo / CM);
  const float* xp = x + (((size_t)b * CM) * NN + n) * Tin + t;
  const size_t cs = (size_t)NN * Tin;
  float f = fb[o], g = gb[o];
#pragma unroll
  for (int c = 0; c < CM; ++c) {
    float x0 = xp[(size_t)c * cs], x1 = xp[(size_t)c * cs + 1];
    f += fw[(o * CM + c) * 2] * x0 + fw[(o * CM + c) * 2 + 1] * x1;
    g += gw[(o * CM + c) * 2] * x0 + gw[(o * CM + c) * 2 + 1] * x1;
  }
  xg[idx] = tanhf(f) * (1.0f / (1.0f + __expf(-g)));
}

// ---------------------------------------------------------------------------
// Pack WMMA left operand: f32 [B][C][N][T] -> bf16 [B][C][T][N]
// (row m = c*T + t is then contiguous along K = node index)
// ---------------------------------------------------------------------------
__global__ void k_pack_bf16(const float* __restrict__ x, __bf16* __restrict__ xbf, int T) {
  size_t idx = (size_t)blockIdx.x * blockDim.x + threadIdx.x;
  const size_t total = (size_t)BQ * CM * (size_t)T * NN;
  if (idx >= total) return;
  int v = (int)(idx % NN);
  size_t r = idx / NN;
  int t = (int)(r % T);
  size_t bc = r / T;
  xbf[idx] = f2bf(x[(bc * NN + v) * T + t]);
}

// ---------------------------------------------------------------------------
// nconv via WMMA bf16: Y[b][c][w][t] = sum_v X[b][c][v][t] * A[b][v][w]
//   left  xbf  [B][C][T][NN]   (M = c*T+t, K = v)
//   right adjT [B][NN(w)][NN(v)]  (N = w, K = v contiguous)
// Each wave: one 16(M) x 64(N) strip, K-loop over 2048 in steps of 32.
// A 16x32 / B 32x16 bf16 use contiguous K (k = 16h + s), not the ISA table order; f32 C/D per CDNA5 ISA 7.12.2.
// ---------------------------------------------------------------------------
__global__ __launch_bounds__(256) void k_nconv_wmma(
    const __bf16* __restrict__ xbf, const __bf16* __restrict__ adjT,
    float* __restrict__ y, int T) {
  const int lane = threadIdx.x & 31;
  const int wave = threadIdx.x >> 5;
  const int h    = lane >> 4;    // lane half
  const int lm   = lane & 15;
  const int b    = blockIdx.z;
  const int m0   = blockIdx.y * 16;               // row tile over M = 32*T
  const int n0   = (blockIdx.x * 8 + wave) * 64;  // 4 n-tiles of 16 per wave

  const __bf16* Ar = xbf  + ((size_t)b * CM * T + (size_t)(m0 + lm)) * NN;
  const __bf16* Br = adjT + ((size_t)b * NN + (size_t)(n0 + lm)) * NN;

  v8f acc0 = {}, acc1 = {}, acc2 = {}, acc3 = {};
  for (int k0 = 0; k0 < NN; k0 += 32) {
    union { v16bf v; struct { v8bf lo, hi; } p; } a;
    a.v = *(const v16bf*)(Ar + k0 + 16 * h);         // K chunk [k0+16h .. +16)
    const int kb = k0 + 16 * h;                      // B: 16 contiguous K per lane
    v16bf b0 = *(const v16bf*)(Br + (size_t)0 * 16 * NN + kb);
    v16bf b1 = *(const v16bf*)(Br + (size_t)1 * 16 * NN + kb);
    v16bf b2 = *(const v16bf*)(Br + (size_t)2 * 16 * NN + kb);
    v16bf b3 = *(const v16bf*)(Br + (size_t)3 * 16 * NN + kb);
    acc0 = __builtin_amdgcn_wmma_f32_16x16x32_bf16(false, a.v, false, b0, (short)0, acc0, false, false);
    acc1 = __builtin_amdgcn_wmma_f32_16x16x32_bf16(false, a.v, false, b1, (short)0, acc1, false, false);
    acc2 = __builtin_amdgcn_wmma_f32_16x16x32_bf16(false, a.v, false, b2, (short)0, acc2, false, false);
    acc3 = __builtin_amdgcn_wmma_f32_16x16x32_bf16(false, a.v, false, b3, (short)0, acc3, false, false);
  }
  union U { v8f v; float f[8]; } d0, d1, d2, d3;
  d0.v = acc0; d1.v = acc1; d2.v = acc2; d3.v = acc3;
#pragma unroll
  for (int r = 0; r < 8; ++r) {
    int M = m0 + r + 8 * h;            // C/D layout: row = vgpr + 8*(lane>=16)
    int c = M / T, t = M - c * T;
    float* yp = y + (((size_t)b * CM + c) * NN) * T + t;
    yp[(size_t)(n0 +  0 + lm) * T] = d0.f[r];
    yp[(size_t)(n0 + 16 + lm) * T] = d1.f[r];
    yp[(size_t)(n0 + 32 + lm) * T] = d2.f[r];
    yp[(size_t)(n0 + 48 + lm) * T] = d3.f[r];
  }
}

// ---------------------------------------------------------------------------
// conv1x1 (32->32 slice of gc_w) accumulate: acc += W_slice @ xin  (+bias init)
// ---------------------------------------------------------------------------
__global__ void k_conv32_acc(const float* __restrict__ xin, const float* __restrict__ W,
                             int ldW, const float* __restrict__ bias,
                             float* __restrict__ acc, int T, int initF) {
  size_t idx = (size_t)blockIdx.x * blockDim.x + threadIdx.x;
  const size_t total = (size_t)BQ * CM * NN * T;
  if (idx >= total) return;
  int t = (int)(idx % T);
  size_t r = idx / T;
  int n = (int)(r % NN);
  size_t bo = r / NN;
  int o = (int)(bo % CM);
  int b = (int)(bo / CM);
  const float* xp = xin + (((size_t)b * CM) * NN + n) * T + t;
  const size_t cs = (size_t)NN * T;
  float s = initF ? bias[o] : acc[idx];
#pragma unroll
  for (int c = 0; c < CM; ++c) s += W[o * ldW + c] * xp[(size_t)c * cs];
  acc[idx] = s;
}

// ---------------------------------------------------------------------------
// x_next = (gc + residual_tail) * gamma/sqrt(1+eps) + beta
// ---------------------------------------------------------------------------
__global__ void k_bn_res(const float* __restrict__ gc, const float* __restrict__ xres,
                         int Tres, const float* __restrict__ gamma,
                         const float* __restrict__ beta, float* __restrict__ xout, int T) {
  size_t idx = (size_t)blockIdx.x * blockDim.x + threadIdx.x;
  const size_t total = (size_t)BQ * CM * NN * T;
  if (idx >= total) return;
  int t = (int)(idx % T);
  size_t r = idx / T;
  int n = (int)(r % NN);
  size_t bo = r / NN;
  int c = (int)(bo % CM);
  int b = (int)(bo / CM);
  float inv = gamma[c] * rsqrtf(1.0f + 1e-5f);
  size_t ri = (((size_t)b * CM + c) * NN + n) * Tres + (size_t)(t + (Tres - T));
  xout[idx] = (gc[idx] + xres[ri]) * inv + beta[c];
}

// ---------------------------------------------------------------------------
// Fused tail: skip = sum_i skip_w[i] @ xg_i(tail) ; relu ; end1 ; relu ; end2
// Block handles (b, t9, 16 columns). LDS-staged matvecs.
// out [B][9][N][24]
// ---------------------------------------------------------------------------
__global__ __launch_bounds__(256) void k_final(
    const float* __restrict__ xg0, const float* __restrict__ xg1,
    const float* __restrict__ xg2, const float* __restrict__ xg3,
    const float* __restrict__ skw, const float* __restrict__ skb,
    const float* __restrict__ e1w, const float* __restrict__ e1b,
    const float* __restrict__ e2w, const float* __restrict__ e2b,
    float* __restrict__ out) {
  __shared__ float xls[4][CM][NCOL];
  __shared__ float sS[NCOL][DFF];
  __shared__ float hS[NCOL][DFF];
  const int tid = threadIdx.x;
  const int b = blockIdx.z, t9 = blockIdx.y, n0 = blockIdx.x * NCOL;
  const float* xg[4] = {xg0, xg1, xg2, xg3};
  // stage 0: load gated-activation tails (layer i has T_i = 12-i; skip tails align at end)
  for (int j = tid; j < 4 * CM * NCOL; j += 256) {
    int i   = j / (CM * NCOL);
    int rem = j % (CM * NCOL);
    int c   = rem / NCOL;
    int col = rem % NCOL;
    int Ti  = 12 - i;
    int t   = Ti - 9 + t9;
    xls[i][c][col] = xg[i][(((size_t)b * CM + c) * NN + (n0 + col)) * Ti + t];
  }
  __syncthreads();
  // stage 1: skip projection (4 x 256x32) + relu
  {
    int o = tid;
    float acc[NCOL];
    float bs = skb[o] + skb[DFF + o] + skb[2 * DFF + o] + skb[3 * DFF + o];
#pragma unroll
    for (int col = 0; col < NCOL; ++col) acc[col] = bs;
    for (int i = 0; i < 4; ++i)
      for (int c = 0; c < CM; ++c) {
        float wv = skw[((size_t)i * DFF + o) * CM + c];
#pragma unroll
        for (int col = 0; col < NCOL; ++col) acc[col] += wv * xls[i][c][col];
      }
#pragma unroll
    for (int col = 0; col < NCOL; ++col) sS[col][o] = fmaxf(acc[col], 0.0f);
  }
  __syncthreads();
  // stage 2: end1 (256x256) + relu
  {
    int o = tid;
    float acc[NCOL];
#pragma unroll
    for (int col = 0; col < NCOL; ++col) acc[col] = e1b[o];
    for (int c = 0; c < DFF; ++c) {
      float wv = e1w[(size_t)o * DFF + c];
#pragma unroll
      for (int col = 0; col < NCOL; ++col) acc[col] += wv * sS[col][c];
    }
#pragma unroll
    for (int col = 0; col < NCOL; ++col) hS[col][o] = fmaxf(acc[col], 0.0f);
  }
  __syncthreads();
  // stage 3: end2 (24x256) + transpose-store
  for (int j = tid; j < 24 * NCOL; j += 256) {
    int o = j % 24, col = j / 24;
    float acc = e2b[o];
    for (int c = 0; c < DFF; ++c) acc += e2w[o * DFF + c] * hS[col][c];
    out[(((size_t)b * 9 + t9) * NN + (n0 + col)) * 24 + o] = acc;
  }
}

// ---------------------------------------------------------------------------
extern "C" void kernel_launch(void* const* d_in, const int* in_sizes, int n_in,
                              void* d_out, int out_size, void* d_ws, size_t ws_size,
                              hipStream_t stream) {
  (void)in_sizes; (void)n_in; (void)out_size; (void)ws_size;
  const float* x       = (const float*)d_in[0];
  const float* adj0    = (const float*)d_in[1];
  const float* adj1    = (const float*)d_in[2];
  const float* start_w = (const float*)d_in[3];
  const float* start_b = (const float*)d_in[4];
  const float* filt_w  = (const float*)d_in[5];
  const float* filt_b  = (const float*)d_in[6];
  const float* gate_w  = (const float*)d_in[7];
  const float* gate_b  = (const float*)d_in[8];
  const float* skip_w  = (const float*)d_in[9];
  const float* skip_b  = (const float*)d_in[10];
  const float* gc_w    = (const float*)d_in[11];
  const float* gc_b    = (const float*)d_in[12];
  const float* bn_g    = (const float*)d_in[13];
  const float* bn_b    = (const float*)d_in[14];
  const float* e1w     = (const float*)d_in[15];
  const float* e1b     = (const float*)d_in[16];
  const float* e2w     = (const float*)d_in[17];
  const float* e2b     = (const float*)d_in[18];
  float* out = (float*)d_out;

  // workspace layout
  char* w = (char*)d_ws;
  const size_t SZ_X13 = (size_t)BQ * CM * NN * 13 * sizeof(float);
  const size_t SZ_X12 = (size_t)BQ * CM * NN * 12 * sizeof(float);
  const size_t SZ_XBF = (size_t)BQ * CM * 12 * NN * 2;
  const size_t SZ_ADJ = (size_t)BQ * NN * NN * 2;
  size_t off = 0;
  float* xA = (float*)(w + off); off += SZ_X13;
  float* xB = (float*)(w + off); off += SZ_X13;
  float* xg[4];
  for (int i = 0; i < 4; ++i) { xg[i] = (float*)(w + off); off += SZ_X12; }
  float*  t1  = (float*)(w + off);  off += SZ_X12;
  float*  t2  = (float*)(w + off);  off += SZ_X12;
  float*  gc  = (float*)(w + off);  off += SZ_X12;
  __bf16* xbf = (__bf16*)(w + off); off += SZ_XBF;
  __bf16* aT0 = (__bf16*)(w + off); off += SZ_ADJ;
  __bf16* aT1 = (__bf16*)(w + off); off += SZ_ADJ;

  dim3 tb(256);
  {
    dim3 g(NN / 32, NN / 32, BQ);
    k_adj_t_bf16<<<g, tb, 0, stream>>>(adj0, aT0);
    k_adj_t_bf16<<<g, tb, 0, stream>>>(adj1, aT1);
  }
  {
    size_t total = (size_t)BQ * CM * NN * 13;
    k_start<<<dim3((unsigned)((total + 255) / 256)), tb, 0, stream>>>(x, start_w, start_b, xA);
  }
  float* cur = xA; float* nxt = xB;
  int Tcur = 13;
  for (int i = 0; i < 4; ++i) {
    int Tout = Tcur - 1;
    size_t totO = (size_t)BQ * CM * NN * Tout;
    dim3 ge((unsigned)((totO + 255) / 256));
    k_filt_gate<<<ge, tb, 0, stream>>>(cur, filt_w + (size_t)i * CM * CM * 2, filt_b + i * CM,
                                       gate_w + (size_t)i * CM * CM * 2, gate_b + i * CM,
                                       xg[i], Tcur);
    if (i == 3) break;
    const float* gw = gc_w + (size_t)i * CM * 160;
    dim3 gn(4, 2 * Tout, BQ);
    // concat slices: [xg, A0 xg, A0^2 xg, A1 xg, A1^2 xg] folded into gc accumulator
    k_conv32_acc<<<ge, tb, 0, stream>>>(xg[i], gw + 0, 160, gc_b + i * CM, gc, Tout, 1);
    k_pack_bf16<<<ge, tb, 0, stream>>>(xg[i], xbf, Tout);
    k_nconv_wmma<<<gn, tb, 0, stream>>>(xbf, aT0, t1, Tout);
    k_conv32_acc<<<ge, tb, 0, stream>>>(t1, gw + 32, 160, nullptr, gc, Tout, 0);
    k_nconv_wmma<<<gn, tb, 0, stream>>>(xbf, aT1, t2, Tout);
    k_conv32_acc<<<ge, tb, 0, stream>>>(t2, gw + 96, 160, nullptr, gc, Tout, 0);
    k_pack_bf16<<<ge, tb, 0, stream>>>(t1, xbf, Tout);
    k_nconv_wmma<<<gn, tb, 0, stream>>>(xbf, aT0, t1, Tout);
    k_conv32_acc<<<ge, tb, 0, stream>>>(t1, gw + 64, 160, nullptr, gc, Tout, 0);
    k_pack_bf16<<<ge, tb, 0, stream>>>(t2, xbf, Tout);
    k_nconv_wmma<<<gn, tb, 0, stream>>>(xbf, aT1, t2, Tout);
    k_conv32_acc<<<ge, tb, 0, stream>>>(t2, gw + 128, 160, nullptr, gc, Tout, 0);
    k_bn_res<<<ge, tb, 0, stream>>>(gc, cur, Tcur, bn_g + i * CM, bn_b + i * CM, nxt, Tout);
    float* tmp = cur; cur = nxt; nxt = tmp;
    Tcur = Tout;
  }
  {
    dim3 g(NN / NCOL, 9, BQ);
    k_final<<<g, tb, 0, stream>>>(xg[0], xg[1], xg[2], xg[3],
                                  skip_w, skip_b, e1w, e1b, e2w, e2b, out);
  }
}
